// SiglipAttention_91199335563717
// MI455X (gfx1250) — matrix operand order repaired and checked by reading, not compiled, not run on hardware
//
#include <hip/hip_runtime.h>
#include <hip/hip_bf16.h>
#include <stdint.h>

typedef __bf16 bf16;
typedef __attribute__((ext_vector_type(8)))  bf16  v8bf;
typedef __attribute__((ext_vector_type(16))) bf16  v16bf;
typedef __attribute__((ext_vector_type(8)))  float v8f;

union BFrag { v16bf v; v8bf h[2]; };

#define HID   1152
#define HEADS 16
#define HDIM  72
#define HPAD  96
#define BB    2
#define TT    4096
#define ROWS  (BB*TT)

// ---- gfx1250 async global->LDS path via inline asm (VFLAT: vdst = per-lane LDS
// byte address, vaddr = 64-bit global address). Falls back to sync staging. ----
#if defined(__gfx1250__)
#define HAS_ASYNC_LDS 1
#endif

// `ldsoff` = LDS byte offset of destination, `g` = global source (16 bytes/lane)
static __device__ __forceinline__ void copy16_to_lds(const bf16* g, uint32_t ldsoff,
                                                     bf16* l_fallback) {
#ifdef HAS_ASYNC_LDS
  asm volatile("global_load_async_to_lds_b128 %0, %1, off"
               :
               : "v"(ldsoff), "v"(g)
               : "memory");
  (void)l_fallback;
#else
  (void)ldsoff;
  *(v8bf*)l_fallback = *(const v8bf*)g;
#endif
}

static __device__ __forceinline__ void wait_async_lds() {
#ifdef HAS_ASYNC_LDS
#if __has_builtin(__builtin_amdgcn_s_wait_asynccnt)
  __builtin_amdgcn_s_wait_asynccnt(0);
#else
  asm volatile("s_wait_asynccnt 0" ::: "memory");
#endif
#endif
}

static __device__ __forceinline__ v8f v8f_zero() {
  v8f z = {0.f,0.f,0.f,0.f,0.f,0.f,0.f,0.f};
  return z;
}

static __device__ __forceinline__ v8f wmma_bf16(const BFrag& a, const BFrag& b, v8f c) {
  return __builtin_amdgcn_wmma_f32_16x16x32_bf16(false, a.v, false, b.v, (short)0, c, false, false);
}

// ---------------- conversion kernels ----------------

__global__ void k_cvt_x(const float* __restrict__ x, bf16* __restrict__ xb) {
  size_t i = (size_t)blockIdx.x * 256 + threadIdx.x;
  xb[i] = (bf16)x[i];
}

// w[k][n] f32 -> wT[n][k] bf16   (B-fragments want contiguous K per output column)
__global__ void k_cvt_wT(const float* __restrict__ wq, const float* __restrict__ wk,
                         const float* __restrict__ wv, const float* __restrict__ wo,
                         bf16* __restrict__ tq, bf16* __restrict__ tk,
                         bf16* __restrict__ tv, bf16* __restrict__ to_) {
  const int z = blockIdx.z;
  const float* w = (z == 0) ? wq : (z == 1) ? wk : (z == 2) ? wv : wo;
  bf16*        t = (z == 0) ? tq : (z == 1) ? tk : (z == 2) ? tv : to_;
  int i = blockIdx.x * 256 + threadIdx.x;
  int n = i / HID, k = i % HID;
  t[(size_t)n * HID + k] = (bf16)w[(size_t)k * HID + n];
}

__global__ void k_zero(uint32_t* __restrict__ p, size_t nwords) {
  size_t i = (size_t)blockIdx.x * blockDim.x + threadIdx.x;
  size_t stride = (size_t)gridDim.x * blockDim.x;
  for (; i < nwords; i += stride) p[i] = 0u;
}

// ---------------- fused QKV projection GEMM ----------------
// grid: (ROWS/128, HID/128, 3)  block: 256 (8 waves); wave tile 32x64 (2x4 WMMA, 8 acc)
__global__ __launch_bounds__(256) void k_qkv(
    const bf16* __restrict__ xb,
    const bf16* __restrict__ wqT, const bf16* __restrict__ wkT, const bf16* __restrict__ wvT,
    const float* __restrict__ bq, const float* __restrict__ bk, const float* __restrict__ bv,
    bf16* __restrict__ Qp, bf16* __restrict__ Kp, bf16* __restrict__ Vt)
{
  const int z = blockIdx.z;
  const bf16*  wT   = (z == 0) ? wqT : (z == 1) ? wkT : wvT;
  const float* bias = (z == 0) ? bq  : (z == 1) ? bk  : bv;

  const int lane = threadIdx.x & 31;
  const int w    = threadIdx.x >> 5;
  const int l15  = lane & 15, lh = lane >> 4;
  const int row0 = blockIdx.x * 128 + (w & 3) * 32;
  const int col0 = blockIdx.y * 128 + (w >> 2) * 64;

  v8f acc[2][4];
#pragma unroll
  for (int mi = 0; mi < 2; ++mi)
#pragma unroll
    for (int ni = 0; ni < 4; ++ni) acc[mi][ni] = v8f_zero();

  for (int k0 = 0; k0 < HID; k0 += 32) {
    BFrag a[2], b[4];
#pragma unroll
    for (int mi = 0; mi < 2; ++mi) {                       // A: 16x32, row = l15
      const bf16* p = xb + (size_t)(row0 + 16 * mi + l15) * HID + k0 + 8 * lh;
      a[mi].h[0] = *(const v8bf*)p;
      a[mi].h[1] = *(const v8bf*)(p + 16);
    }
#pragma unroll
    for (int ni = 0; ni < 4; ++ni) {                       // B: 32x16, col = l15
      const bf16* p = wT + (size_t)(col0 + 16 * ni + l15) * HID + k0 + 8 * lh;
      b[ni].h[0] = *(const v8bf*)p;
      b[ni].h[1] = *(const v8bf*)(p + 16);
    }
#pragma unroll
    for (int mi = 0; mi < 2; ++mi)
#pragma unroll
      for (int ni = 0; ni < 4; ++ni)
        acc[mi][ni] = wmma_bf16(a[mi], b[ni], acc[mi][ni]);
  }

#pragma unroll
  for (int mi = 0; mi < 2; ++mi)
#pragma unroll
    for (int ni = 0; ni < 4; ++ni)
#pragma unroll
      for (int j = 0; j < 8; ++j) {
        const int r   = row0 + 16 * mi + 8 * lh + j;       // D: M = j + 8*(lane>>4)
        const int col = col0 + 16 * ni + l15;              //    N = lane&15
        const float val = acc[mi][ni][j] + bias[col];
        const int bidx = r >> 12, t = r & (TT - 1);
        const int h = col / HDIM, c = col % HDIM;
        if (z == 2) {
          Vt[((size_t)(bidx * HEADS + h) * HPAD + c) * TT + t] = (bf16)val;   // [B][H][HPAD][T]
        } else {
          bf16* P = z ? Kp : Qp;                                              // [B][H][T][HPAD]
          P[((size_t)(bidx * HEADS + h) * TT + t) * HPAD + c] = (bf16)val;
        }
      }
}

// ---------------- flash attention ----------------
// grid: (TT/128, HEADS, BB)  block: 256; each wave owns 16 q-rows.
// K/V kv-chunks (32 wide) staged in LDS once per block via double-buffered
// async global->LDS DMA; all 8 waves consume the same staged tiles.
struct __align__(16) AttnSmem {
  bf16 Ks[2][32][104];  // 32 kv rows x 96 (+8 pad) -> 208B stride
  bf16 Vs[2][96][40];   // 96 cols x 32 kv (+8 pad) -> 80B stride
  bf16 pbuf[8][16][40]; // per-wave 16x32 P tile, 80B stride
};

__global__ __launch_bounds__(256) void k_attn(
    const bf16* __restrict__ Qp, const bf16* __restrict__ Kp,
    const bf16* __restrict__ Vt, bf16* __restrict__ ctx)
{
  __shared__ AttnSmem sm;
  const char* smbase = (const char*)&sm;   // single static LDS alloc -> offset 0

  const int tid  = threadIdx.x;
  const int lane = tid & 31;
  const int w    = tid >> 5;
  const int l15  = lane & 15, lh = lane >> 4;
  const int b = blockIdx.z, h = blockIdx.y;
  const int q0 = blockIdx.x * 128 + w * 16;

  const bf16* Qh = Qp + (size_t)(b * HEADS + h) * TT * HPAD;
  const bf16* Kh = Kp + (size_t)(b * HEADS + h) * TT * HPAD;
  const bf16* Vh = Vt + (size_t)(b * HEADS + h) * HPAD * TT;

  auto ldsoff = [&](const bf16* p) -> uint32_t {
    return (uint32_t)((const char*)p - smbase);
  };

  // stage one kv chunk (K: 384 16B chunks, V: 384 16B chunks) into buffer `buf`
  auto stage = [&](int kv, int buf) {
    {
      int r = tid / 12, q = tid % 12;                  // K row = kv idx, 12 chunks/row
      bf16* d = &sm.Ks[buf][r][q * 8];
      copy16_to_lds(Kh + (size_t)(kv + r) * HPAD + q * 8, ldsoff(d), d);
      if (tid < 128) {
        int c2 = tid + 256;
        int r2 = c2 / 12, q2 = c2 % 12;
        bf16* d2 = &sm.Ks[buf][r2][q2 * 8];
        copy16_to_lds(Kh + (size_t)(kv + r2) * HPAD + q2 * 8, ldsoff(d2), d2);
      }
    }
    {
      int r = tid >> 2, q = tid & 3;                   // V row = head col, 4 chunks/row
      bf16* d = &sm.Vs[buf][r][q * 8];
      copy16_to_lds(Vh + (size_t)r * TT + kv + q * 8, ldsoff(d), d);
      if (tid < 128) {
        int c2 = tid + 256;
        int r2 = c2 >> 2, q2 = c2 & 3;
        bf16* d2 = &sm.Vs[buf][r2][q2 * 8];
        copy16_to_lds(Vh + (size_t)r2 * TT + kv + q2 * 8, ldsoff(d2), d2);
      }
    }
  };

  // Q fragments (A layout): 3 K-steps covering padded head dim 96
  BFrag qf[3];
#pragma unroll
  for (int ks = 0; ks < 3; ++ks) {
    const bf16* p = Qh + (size_t)(q0 + l15) * HPAD + 32 * ks + 8 * lh;
    qf[ks].h[0] = *(const v8bf*)p;
    qf[ks].h[1] = *(const v8bf*)(p + 16);
  }

  v8f o[6];
#pragma unroll
  for (int n = 0; n < 6; ++n) o[n] = v8f_zero();
  float m[8], l[8];
#pragma unroll
  for (int j = 0; j < 8; ++j) { m[j] = -3.0e38f; l[j] = 0.0f; }

  const float sc = 0.11785113019f * 1.44269504089f;   // (1/sqrt(72)) * log2(e)

  stage(0, 0);

  for (int kv0 = 0; kv0 < TT; kv0 += 32) {
    const int cur = (kv0 >> 5) & 1;
    wait_async_lds();
    __syncthreads();                                  // staged tiles visible to all waves
    if (kv0 + 32 < TT) stage(kv0 + 32, cur ^ 1);      // overlap next DMA with compute

    // S = Q * K^T for this 16x32 kv chunk (two 16x16 D tiles)
    v8f s[2];
#pragma unroll
    for (int ni = 0; ni < 2; ++ni) {
      s[ni] = v8f_zero();
#pragma unroll
      for (int ks = 0; ks < 3; ++ks) {
        BFrag kf;                                     // B layout from LDS: col = kv index
        const bf16* p = &sm.Ks[cur][16 * ni + l15][32 * ks + 8 * lh];
        kf.h[0] = *(const v8bf*)p;
        kf.h[1] = *(const v8bf*)(p + 16);
        s[ni] = wmma_bf16(qf[ks], kf, s[ni]);
      }
    }

    // base-2 scaling + row max (row M=j+8*lh lives in VGPR j across 16 lanes of a half)
    float rmax[8];
#pragma unroll
    for (int j = 0; j < 8; ++j) {
      float a0 = s[0][j] * sc, a1 = s[1][j] * sc;
      s[0][j] = a0; s[1][j] = a1;
      rmax[j] = fmaxf(a0, a1);
    }
#pragma unroll
    for (int off = 1; off < 16; off <<= 1)
#pragma unroll
      for (int j = 0; j < 8; ++j)
        rmax[j] = fmaxf(rmax[j], __shfl_xor(rmax[j], off, 32));

    float corr[8];
#pragma unroll
    for (int j = 0; j < 8; ++j) {
      float mn = fmaxf(m[j], rmax[j]);
      corr[j] = __builtin_exp2f(m[j] - mn);
      m[j] = mn;
      l[j] *= corr[j];
    }

    float rsum[8];
#pragma unroll
    for (int j = 0; j < 8; ++j) {
      float p0 = __builtin_exp2f(s[0][j] - m[j]);
      float p1 = __builtin_exp2f(s[1][j] - m[j]);
      s[0][j] = p0; s[1][j] = p1;
      rsum[j] = p0 + p1;
    }
#pragma unroll
    for (int off = 1; off < 16; off <<= 1)
#pragma unroll
      for (int j = 0; j < 8; ++j)
        rsum[j] += __shfl_xor(rsum[j], off, 32);
#pragma unroll
    for (int j = 0; j < 8; ++j) l[j] += rsum[j];

#pragma unroll
    for (int n = 0; n < 6; ++n)
#pragma unroll
      for (int j = 0; j < 8; ++j) o[n][j] *= corr[j];

    // D-layout -> A-layout for P via per-wave LDS slab
#pragma unroll
    for (int ni = 0; ni < 2; ++ni)
#pragma unroll
      for (int j = 0; j < 8; ++j)
        sm.pbuf[w][j + 8 * lh][16 * ni + l15] = (bf16)s[ni][j];
    __syncthreads();

    BFrag pa;
    {
      const bf16* pp = &sm.pbuf[w][l15][8 * lh];
      pa.h[0] = *(const v8bf*)pp;
      pa.h[1] = *(const v8bf*)(pp + 16);
    }

    // O += P * V  (V staged transposed: contiguous along kv)
#pragma unroll
    for (int n = 0; n < 6; ++n) {
      BFrag vf;
      const bf16* p = &sm.Vs[cur][16 * n + l15][8 * lh];
      vf.h[0] = *(const v8bf*)p;
      vf.h[1] = *(const v8bf*)(p + 16);
      o[n] = wmma_bf16(pa, vf, o[n]);
    }

    __syncthreads();   // all waves done reading buf[cur] before it is re-staged
  }

  float inv[8];
#pragma unroll
  for (int j = 0; j < 8; ++j) inv[j] = 1.0f / l[j];

#pragma unroll
  for (int n = 0; n < 6; ++n)
#pragma unroll
    for (int j = 0; j < 8; ++j) {
      const int col = 16 * n + l15;
      if (col < HDIM) {
        const int t = q0 + 8 * lh + j;
        ctx[((size_t)b * TT + t) * HID + h * HDIM + col] = (bf16)(o[n][j] * inv[j]);
      }
    }
}

// ---------------- output projection GEMM ----------------
// grid: (ROWS/128, HID/128)  block: 256; wave tile 32x64
__global__ __launch_bounds__(256) void k_out(
    const bf16* __restrict__ ctx, const bf16* __restrict__ woT,
    const float* __restrict__ bo, float* __restrict__ out)
{
  const int lane = threadIdx.x & 31;
  const int w    = threadIdx.x >> 5;
  const int l15  = lane & 15, lh = lane >> 4;
  const int row0 = blockIdx.x * 128 + (w & 3) * 32;
  const int col0 = blockIdx.y * 128 + (w >> 2) * 64;

  v8f acc[2][4];
#pragma unroll
  for (int mi = 0; mi < 2; ++mi)
#pragma unroll
    for (int ni = 0; ni < 4; ++ni) acc[mi][ni] = v8f_zero();

  for (int k0 = 0; k0 < HID; k0 += 32) {
    BFrag a[2], b[4];
#pragma unroll
    for (int mi = 0; mi < 2; ++mi) {
      const bf16* p = ctx + (size_t)(row0 + 16 * mi + l15) * HID + k0 + 8 * lh;
      a[mi].h[0] = *(const v8bf*)p;
      a[mi].h[1] = *(const v8bf*)(p + 16);
    }
#pragma unroll
    for (int ni = 0; ni < 4; ++ni) {
      const bf16* p = woT + (size_t)(col0 + 16 * ni + l15) * HID + k0 + 8 * lh;
      b[ni].h[0] = *(const v8bf*)p;
      b[ni].h[1] = *(const v8bf*)(p + 16);
    }
#pragma unroll
    for (int mi = 0; mi < 2; ++mi)
#pragma unroll
      for (int ni = 0; ni < 4; ++ni)
        acc[mi][ni] = wmma_bf16(a[mi], b[ni], acc[mi][ni]);
  }

#pragma unroll
  for (int mi = 0; mi < 2; ++mi)
#pragma unroll
    for (int ni = 0; ni < 4; ++ni)
#pragma unroll
      for (int j = 0; j < 8; ++j) {
        const int r   = row0 + 16 * mi + 8 * lh + j;
        const int col = col0 + 16 * ni + l15;
        out[(size_t)r * HID + col] = acc[mi][ni][j] + bo[col];
      }
}

// ---------------- launch ----------------
extern "C" void kernel_launch(void* const* d_in, const int* in_sizes, int n_in,
                              void* d_out, int out_size, void* d_ws, size_t ws_size,
                              hipStream_t stream) {
  const float* x  = (const float*)d_in[0];
  const float* wq = (const float*)d_in[1];
  const float* bq = (const float*)d_in[2];
  const float* wk = (const float*)d_in[3];
  const float* bk = (const float*)d_in[4];
  const float* wv = (const float*)d_in[5];
  const float* bv = (const float*)d_in[6];
  const float* wo = (const float*)d_in[7];
  const float* bo = (const float*)d_in[8];
  float* out = (float*)d_out;

  char* ws = (char*)d_ws;
  size_t off = 0;
  auto alloc = [&](size_t bytes) -> void* {
    void* p = ws + off;
    off += (bytes + 255) & ~(size_t)255;
    return p;
  };

  bf16* xb  = (bf16*)alloc((size_t)ROWS * HID * 2);
  bf16* wqT = (bf16*)alloc((size_t)HID * HID * 2);
  bf16* wkT = (bf16*)alloc((size_t)HID * HID * 2);
  bf16* wvT = (bf16*)alloc((size_t)HID * HID * 2);
  bf16* woT = (bf16*)alloc((size_t)HID * HID * 2);
  bf16* Qp  = (bf16*)alloc((size_t)BB * HEADS * TT * HPAD * 2);
  bf16* Kp  = (bf16*)alloc((size_t)BB * HEADS * TT * HPAD * 2);
  bf16* Vt  = (bf16*)alloc((size_t)BB * HEADS * TT * HPAD * 2);
  bf16* ctx = (bf16*)alloc((size_t)ROWS * HID * 2);

  k_cvt_x<<<(ROWS * HID) / 256, 256, 0, stream>>>(x, xb);

  dim3 gw((HID * HID) / 256, 1, 4);
  k_cvt_wT<<<gw, 256, 0, stream>>>(wq, wk, wv, wo, wqT, wkT, wvT, woT);

  size_t zwords = ((size_t)BB * HEADS * TT * HPAD * 2 * 3) / 4;
  k_zero<<<4096, 256, 0, stream>>>((uint32_t*)Qp, zwords);

  dim3 gq(ROWS / 128, HID / 128, 3);
  k_qkv<<<gq, 256, 0, stream>>>(xb, wqT, wkT, wvT, bq, bk, bv, Qp, Kp, Vt);

  dim3 ga(TT / 128, HEADS, BB);
  k_attn<<<ga, 256, 0, stream>>>(Qp, Kp, Vt, ctx);

  dim3 go(ROWS / 128, HID / 128, 1);
  k_out<<<go, 256, 0, stream>>>(ctx, woT, bo, out);
}
